// MultiScaleBlock_48163763258082
// MI455X (gfx1250) — matrix operand order repaired and checked by reading, not compiled, not run on hardware
//
#include <hip/hip_runtime.h>

// ---------------- types ----------------
typedef __bf16 bf16;
typedef bf16  v16bf __attribute__((ext_vector_type(16)));
typedef bf16  v8bf  __attribute__((ext_vector_type(8)));
typedef float v8f   __attribute__((ext_vector_type(8)));

// Optional CDNA5 async global->LDS path (guarded so absence never breaks compile)
#if __has_builtin(__builtin_amdgcn_global_load_async_to_lds_b128) && \
    __has_builtin(__builtin_amdgcn_s_wait_asynccnt)
#define USE_ASYNC_LDS 1
typedef int v4i_ld __attribute__((vector_size(16)));  // matches builtin param type
#else
#define USE_ASYNC_LDS 0
#endif

// ---------------- problem constants ----------------
constexpr int BATCH = 4, Himg = 256, Wimg = 256;
constexpr int DIM = 96, DIM_OUT = 192, HD = 64;
constexpr int WS = 8, NWIN = 32;                 // 32x32 windows per image
constexpr int NWINDOWS = BATCH * NWIN * NWIN;    // 4096
constexpr int HP = 128, WP = 128;                // pooled spatial
constexpr float EPS = 1e-6f;

// ---------------- fragment helpers ----------------
__device__ inline v8f wmma_bf16(v16bf a, v16bf b, v8f c) {
  // (neg_a, A, neg_b, B, c_mod, C, reuse_a, reuse_b)
  return __builtin_amdgcn_wmma_f32_16x16x32_bf16(false, a, false, b, (short)0, c,
                                                 false, false);
}

__device__ inline v8f zero8() {
  v8f z;
#pragma unroll
  for (int i = 0; i < 8; ++i) z[i] = 0.0f;
  return z;
}

// A-operand fragment (16x32 bf16) from a row-major [rows][ld] array.
// Contiguous k order (same as B frags): lane m=l&15 holds row; elems i (0..15) -> k = kbase + hi*16 + i
// (hi = lane>>4).
__device__ inline v16bf afrag(const bf16* __restrict__ base, int ld, int row,
                              int kbase, int hi) {
  const bf16* p = base + (size_t)row * ld + kbase + hi * 16;
  v8bf a0 = *(const v8bf*)p;
  v8bf a1 = *(const v8bf*)(p + 8);
  return __builtin_shufflevector(a0, a1, 0, 1, 2, 3, 4, 5, 6, 7,
                                 8, 9, 10, 11, 12, 13, 14, 15);
}

// B-operand fragment from a row-major array where the lane's column is a row
// of the array and the 16 contraction values are contiguous.
// lane l: col = (l&15), elems i -> k = kbase + (l>=16 ? 16 : 0) + i.
__device__ inline v16bf bfrag_rowmajor(const bf16* __restrict__ base, int ld,
                                       int col_row, int kbase, int lane) {
  const bf16* p = base + (size_t)col_row * ld + kbase + ((lane >> 4) << 4);
  return *(const v16bf*)p;
}

// Packed-weight B fragment: ((ntile*(K/32)+ks)*32 + lane)*16 contiguous bf16.
__device__ inline v16bf bfrag_packed(const bf16* __restrict__ wp, int Kdiv32,
                                     int ntile, int ks, int lane) {
  return *(const v16bf*)(wp + (((size_t)ntile * Kdiv32 + ks) * 32 + lane) * 16);
}

// Destination offset in packed layout for source element W[k][n] of a KxN weight.
__device__ inline size_t pk_off(int k, int n, int K) {
  int ntile = n >> 4, ks = k >> 5, kk = k & 31;
  int lane = (n & 15) | ((kk >= 16) ? 16 : 0);
  int i = kk & 15;
  return (((size_t)ntile * (K >> 5) + ks) * 32 + lane) * 16 + i;
}

// ---------------- kernel 0: pack fp32 weight -> bf16 fragment-major ----------------
__global__ __launch_bounds__(256) void pack_w(const float* __restrict__ src,
                                              bf16* __restrict__ dst, int K, int N) {
  int id = blockIdx.x * 256 + threadIdx.x;
  if (id >= K * N) return;
  int k = id / N, n = id % N;
  dst[pk_off(k, n, K)] = (bf16)src[id];
}

// ---------------- kernel 1: LN1 + QKV GEMM + Q-pool + shortcut(proj)+pool ----------------
__global__ __launch_bounds__(256) void k_ln_qkv(
    const float* __restrict__ x, const float* __restrict__ g1,
    const float* __restrict__ b1, const bf16* __restrict__ qkvw_p,
    const float* __restrict__ qkv_b, const bf16* __restrict__ projw_p,
    const float* __restrict__ proj_b, bf16* __restrict__ qp_ws,
    bf16* __restrict__ k_ws, bf16* __restrict__ v_ws, bf16* __restrict__ sc_ws) {
  __shared__ __attribute__((aligned(16))) float xs[64 * 96];  // 24.0 KB
  __shared__ __attribute__((aligned(16))) bf16 xn[64 * 96];   // 12.0 KB
  __shared__ float mu[64], rs[64];

  const int win = blockIdx.x;
  const int wb = win / (NWIN * NWIN);
  const int wr = (win / NWIN) % NWIN;
  const int wc = win % NWIN;
  const int tid = threadIdx.x;

  const float* xw = x + (((size_t)wb * Himg + wr * WS) * Wimg + wc * WS) * DIM;
#if USE_ASYNC_LDS
  // Async DMA of the 8 window rows (each 8 tokens * 96 f32 = 3072B contiguous)
  // straight into LDS: 1536 16B chunks over 256 lanes = 6 issues/lane.
#pragma unroll
  for (int it = 0; it < 6; ++it) {
    int o = (tid + it * 256) * 16;  // byte offset into xs
    int r = o / 3072, w = o % 3072;
    const char* gp = (const char*)(xw + (size_t)r * Wimg * DIM) + w;
    __builtin_amdgcn_global_load_async_to_lds_b128(
        (v4i_ld*)gp, (v4i_ld*)((char*)xs + o), 0, 0);
  }
  __builtin_amdgcn_s_wait_asynccnt(0);
  __syncthreads();
#else
  for (int idx = tid; idx < 64 * 96; idx += 256) {
    int t = idx / 96, d = idx % 96;
    int r = t >> 3, c = t & 7;
    xs[idx] = xw[((size_t)r * Wimg + c) * DIM + d];
  }
  __syncthreads();
#endif
  if (tid < 64) {
    float s = 0.f;
    for (int d = 0; d < 96; ++d) s += xs[tid * 96 + d];
    float m = s * (1.f / 96.f);
    float v = 0.f;
    for (int d = 0; d < 96; ++d) { float t = xs[tid * 96 + d] - m; v += t * t; }
    mu[tid] = m;
    rs[tid] = rsqrtf(v * (1.f / 96.f) + EPS);
  }
  __syncthreads();
  for (int idx = tid; idx < 64 * 96; idx += 256) {
    int t = idx / 96, d = idx % 96;
    xn[idx] = (bf16)((xs[idx] - mu[t]) * rs[t] * g1[d] + b1[d]);
  }
  __syncthreads();

  const int wave = tid >> 5, lane = tid & 31;
  const int lo = lane & 15, hi = lane >> 4;

  // ---- QKV: 64x96 @ 96x576 -> 4 Mtiles x 36 Ntiles ----
  for (int tile = wave; tile < 144; tile += 8) {
    int mt = tile / 36, nt = tile % 36;
    {  // prefetch next tile's packed-weight block (global_prefetch_b8)
      int tn = tile + 8;
      if (tn < 144) __builtin_prefetch(qkvw_p + ((size_t)(tn % 36) * 3) * 512, 0, 0);
    }
    v8f acc = zero8();
#pragma unroll
    for (int ks = 0; ks < 3; ++ks) {
      v16bf a = afrag(xn, 96, mt * 16 + lo, ks * 32, hi);
      v16bf b = bfrag_packed(qkvw_p, 3, nt, ks, lane);
      acc = wmma_bf16(a, b, acc);
    }
    int n = nt * 16 + lo;
    float bias = qkv_b[n];
    if (n < DIM_OUT) {  // ---- Q: 2x2 maxpool straight out of the accumulator
#pragma unroll
      for (int j = 0; j < 4; ++j) {
        float m0 = fmaxf(acc[2 * j], acc[2 * j + 1]);
        float m1 = __shfl_xor(m0, 16, 32);
        float mv = fmaxf(m0, m1) + bias;
        if (hi == 0)
          qp_ws[((size_t)win * 16 + mt * 4 + j) * DIM_OUT + n] = (bf16)mv;
      }
    } else if (n < 2 * DIM_OUT) {  // ---- K: row-major [64][192]
#pragma unroll
      for (int e = 0; e < 8; ++e)
        k_ws[((size_t)win * 64 + mt * 16 + e + 8 * hi) * DIM_OUT + (n - DIM_OUT)] =
            (bf16)(acc[e] + bias);
    } else {  // ---- V: transposed [192][64]; rows of acc are contiguous tokens
      v8bf vv;
#pragma unroll
      for (int e = 0; e < 8; ++e) vv[e] = (bf16)(acc[e] + bias);
      *(v8bf*)(v_ws + ((size_t)win * DIM_OUT + (n - 2 * DIM_OUT)) * 64 + mt * 16 +
               8 * hi) = vv;
    }
  }

  // ---- shortcut: 64x96 @ 96x192, pooled 2x2 -> [16][192] bf16 ----
  for (int tile = wave; tile < 48; tile += 8) {
    int mt = tile / 12, nt = tile % 12;
    v8f acc = zero8();
#pragma unroll
    for (int ks = 0; ks < 3; ++ks) {
      v16bf a = afrag(xn, 96, mt * 16 + lo, ks * 32, hi);
      v16bf b = bfrag_packed(projw_p, 3, nt, ks, lane);
      acc = wmma_bf16(a, b, acc);
    }
    int n = nt * 16 + lo;
    float bias = proj_b[n];
#pragma unroll
    for (int j = 0; j < 4; ++j) {
      float m0 = fmaxf(acc[2 * j], acc[2 * j + 1]);
      float m1 = __shfl_xor(m0, 16, 32);
      float mv = fmaxf(m0, m1) + bias;
      if (hi == 0)
        sc_ws[((size_t)win * 16 + mt * 4 + j) * DIM_OUT + n] = (bf16)mv;
    }
  }
}

// ---------------- kernel 2: attention + attn_proj + residual -> y ----------------
__global__ __launch_bounds__(256) void k_attn(
    const bf16* __restrict__ qp_ws, const bf16* __restrict__ k_ws,
    const bf16* __restrict__ v_ws, const bf16* __restrict__ sc_ws,
    const bf16* __restrict__ attnw_p, const float* __restrict__ attn_b,
    float* __restrict__ y_out) {
  __shared__ __attribute__((aligned(16))) float sc_s[3 * 16 * 64];  // 12 KB
  __shared__ __attribute__((aligned(16))) bf16 pr_s[3 * 16 * 64];   // 6 KB
  __shared__ __attribute__((aligned(16))) bf16 out_s[16 * 192];     // 6 KB

  const int win = blockIdx.x, tid = threadIdx.x;
  const int wave = tid >> 5, lane = tid & 31, lo = lane & 15, hi = lane >> 4;
  const bf16* qpw = qp_ws + (size_t)win * 16 * DIM_OUT;
  const bf16* kw = k_ws + (size_t)win * 64 * DIM_OUT;
  const bf16* vw = v_ws + (size_t)win * DIM_OUT * 64;

  // scores[h][q][key] = (Q . K) * 1/sqrt(64)
  for (int tile = wave; tile < 12; tile += 8) {
    int h = tile >> 2, nt = tile & 3;
    v8f acc = zero8();
#pragma unroll
    for (int ks = 0; ks < 2; ++ks) {
      v16bf a = afrag(qpw, DIM_OUT, lo, h * HD + ks * 32, hi);
      v16bf b = bfrag_rowmajor(kw, DIM_OUT, nt * 16 + lo, h * HD + ks * 32, lane);
      acc = wmma_bf16(a, b, acc);
    }
#pragma unroll
    for (int e = 0; e < 8; ++e)
      sc_s[(h * 16 + e + 8 * hi) * 64 + nt * 16 + lo] = acc[e] * 0.125f;
  }
  __syncthreads();
  if (tid < 48) {  // softmax over 64 keys, one row per thread
    int base = tid * 64;
    float mx = -3.4e38f;
    for (int k = 0; k < 64; ++k) mx = fmaxf(mx, sc_s[base + k]);
    float s = 0.f;
    for (int k = 0; k < 64; ++k) {
      float e = __expf(sc_s[base + k] - mx);
      sc_s[base + k] = e;
      s += e;
    }
    float inv = 1.f / s;
    for (int k = 0; k < 64; ++k) pr_s[base + k] = (bf16)(sc_s[base + k] * inv);
  }
  __syncthreads();
  // out = probs @ V
  for (int tile = wave; tile < 12; tile += 8) {
    int h = tile >> 2, nt = tile & 3;
    v8f acc = zero8();
#pragma unroll
    for (int ks = 0; ks < 2; ++ks) {
      v16bf a = afrag(pr_s + h * 16 * 64, 64, lo, ks * 32, hi);
      v16bf b = bfrag_rowmajor(vw, 64, h * HD + nt * 16 + lo, ks * 32, lane);
      acc = wmma_bf16(a, b, acc);
    }
#pragma unroll
    for (int e = 0; e < 8; ++e)
      out_s[(e + 8 * hi) * DIM_OUT + h * HD + nt * 16 + lo] = (bf16)acc[e];
  }
  __syncthreads();
  // attn_proj (K=192) + bias + shortcut -> y
  const int wb = win / (NWIN * NWIN);
  const int wr = (win / NWIN) % NWIN;
  const int wc = win % NWIN;
  for (int nt = wave; nt < 12; nt += 8) {
    v8f acc = zero8();
#pragma unroll
    for (int ks = 0; ks < 6; ++ks) {
      v16bf a = afrag(out_s, DIM_OUT, lo, ks * 32, hi);
      v16bf b = bfrag_packed(attnw_p, 6, nt, ks, lane);
      acc = wmma_bf16(a, b, acc);
    }
    int n = nt * 16 + lo;
    float bias = attn_b[n];
#pragma unroll
    for (int e = 0; e < 8; ++e) {
      int p = e + 8 * hi;  // pooled token 0..15 -> 4x4 grid
      float yv = acc[e] + bias + (float)sc_ws[((size_t)win * 16 + p) * DIM_OUT + n];
      int gr = wr * 4 + (p >> 2), gc = wc * 4 + (p & 3);
      y_out[(((size_t)wb * HP + gr) * WP + gc) * DIM_OUT + n] = yv;
    }
  }
}

// ---------------- kernel 3: LN2 + MLP (GELU) + residual, in place on d_out ----------------
__global__ __launch_bounds__(256) void k_mlp(
    const float* __restrict__ g2, const float* __restrict__ b2,
    const bf16* __restrict__ w1p, const float* __restrict__ bias1,
    const bf16* __restrict__ w2p, const float* __restrict__ bias2,
    float* __restrict__ y) {
  __shared__ __attribute__((aligned(16))) bf16 yn[64 * 192];  // 24 KB
  __shared__ __attribute__((aligned(16))) bf16 hb[64 * 192];  // 24 KB (hidden chunk)
  __shared__ float mu[64], rs[64];

  const int tid = threadIdx.x;
  const size_t tok0 = (size_t)blockIdx.x * 64;
  if (tid < 64) {
    const float* yr = y + (tok0 + tid) * DIM_OUT;
    float s = 0.f;
    for (int d = 0; d < DIM_OUT; ++d) s += yr[d];
    float m = s * (1.f / DIM_OUT);
    float v = 0.f;
    for (int d = 0; d < DIM_OUT; ++d) { float t = yr[d] - m; v += t * t; }
    mu[tid] = m;
    rs[tid] = rsqrtf(v * (1.f / DIM_OUT) + EPS);
  }
  __syncthreads();
  for (int idx = tid; idx < 64 * DIM_OUT; idx += 256) {
    int t = idx / DIM_OUT, d = idx % DIM_OUT;
    float yv = y[(tok0 + t) * DIM_OUT + d];
    yn[idx] = (bf16)((yv - mu[t]) * rs[t] * g2[d] + b2[d]);
  }
  __syncthreads();

  const int wave = tid >> 5, lane = tid & 31, lo = lane & 15, hi = lane >> 4;
  v8f acc2[6];
#pragma unroll
  for (int s = 0; s < 6; ++s) acc2[s] = zero8();

  for (int cn = 0; cn < 4; ++cn) {  // hidden chunks of 192
    // GEMM1 chunk: yn(64x192) @ w1[:, cn*192:+192], + bias, GELU -> hb
    for (int tile = wave; tile < 48; tile += 8) {
      int mt = tile / 12, nt2 = tile % 12;
      {  // prefetch next tile's packed w1 block
        int tn = tile + 8;
        if (tn < 48)
          __builtin_prefetch(w1p + ((size_t)(cn * 12 + tn % 12) * 6) * 512, 0, 0);
      }
      v8f a1 = zero8();
#pragma unroll
      for (int ks = 0; ks < 6; ++ks) {
        v16bf a = afrag(yn, DIM_OUT, mt * 16 + lo, ks * 32, hi);
        v16bf b = bfrag_packed(w1p, 6, cn * 12 + nt2, ks, lane);
        a1 = wmma_bf16(a, b, a1);
      }
      float bs = bias1[cn * 192 + nt2 * 16 + lo];
#pragma unroll
      for (int e = 0; e < 8; ++e) {
        float xv = a1[e] + bs;
        float ge = 0.5f * xv * (1.f + erff(xv * 0.70710678118f));
        hb[(mt * 16 + e + 8 * hi) * DIM_OUT + nt2 * 16 + lo] = (bf16)ge;
      }
    }
    __syncthreads();
    // GEMM2 partial accumulate: hb(64x192) @ w2[cn*192:+192, :]
#pragma unroll
    for (int s = 0; s < 6; ++s) {
      int tile = wave * 6 + s;
      int mt = tile / 12, nt = tile % 12;
#pragma unroll
      for (int ks = 0; ks < 6; ++ks) {
        v16bf a = afrag(hb, DIM_OUT, mt * 16 + lo, ks * 32, hi);
        v16bf b = bfrag_packed(w2p, 24, nt, cn * 6 + ks, lane);
        acc2[s] = wmma_bf16(a, b, acc2[s]);
      }
    }
    __syncthreads();
  }
#pragma unroll
  for (int s = 0; s < 6; ++s) {
    int tile = wave * 6 + s;
    int mt = tile / 12, nt = tile % 12;
    int n = nt * 16 + lo;
    float bs = bias2[n];
#pragma unroll
    for (int e = 0; e < 8; ++e) {
      size_t row = tok0 + mt * 16 + e + 8 * hi;
      y[row * DIM_OUT + n] = acc2[s][e] + bs + y[row * DIM_OUT + n];
    }
  }
}

// ---------------- host launcher ----------------
extern "C" void kernel_launch(void* const* d_in, const int* in_sizes, int n_in,
                              void* d_out, int out_size, void* d_ws, size_t ws_size,
                              hipStream_t stream) {
  const float* x       = (const float*)d_in[0];
  const float* n1g     = (const float*)d_in[1];
  const float* n1b     = (const float*)d_in[2];
  const float* proj_w  = (const float*)d_in[3];
  const float* proj_b  = (const float*)d_in[4];
  const float* qkv_w   = (const float*)d_in[5];
  const float* qkv_b   = (const float*)d_in[6];
  const float* aproj_w = (const float*)d_in[7];
  const float* aproj_b = (const float*)d_in[8];
  const float* n2g     = (const float*)d_in[9];
  const float* n2b     = (const float*)d_in[10];
  const float* mlp_w1  = (const float*)d_in[11];
  const float* mlp_b1  = (const float*)d_in[12];
  const float* mlp_w2  = (const float*)d_in[13];
  const float* mlp_b2  = (const float*)d_in[14];
  float* y = (float*)d_out;

  char* ws = (char*)d_ws;
  size_t o = 0;
  auto take = [&](size_t bytes) {
    size_t r = o;
    o += (bytes + 1023) & ~(size_t)1023;
    return r;
  };
  size_t o_projw = take((size_t)96 * 192 * 2);
  size_t o_qkvw  = take((size_t)96 * 576 * 2);
  size_t o_attnw = take((size_t)192 * 192 * 2);
  size_t o_w1    = take((size_t)192 * 768 * 2);
  size_t o_w2    = take((size_t)768 * 192 * 2);
  size_t o_qp    = take((size_t)NWINDOWS * 16 * 192 * 2);
  size_t o_k     = take((size_t)NWINDOWS * 64 * 192 * 2);
  size_t o_v     = take((size_t)NWINDOWS * 192 * 64 * 2);
  size_t o_sc    = take((size_t)NWINDOWS * 16 * 192 * 2);
  (void)ws_size; (void)in_sizes; (void)n_in; (void)out_size;

  bf16* projw_p = (bf16*)(ws + o_projw);
  bf16* qkvw_p  = (bf16*)(ws + o_qkvw);
  bf16* attnw_p = (bf16*)(ws + o_attnw);
  bf16* w1p     = (bf16*)(ws + o_w1);
  bf16* w2p     = (bf16*)(ws + o_w2);
  bf16* qp_ws   = (bf16*)(ws + o_qp);
  bf16* k_ws    = (bf16*)(ws + o_k);
  bf16* v_ws    = (bf16*)(ws + o_v);
  bf16* sc_ws   = (bf16*)(ws + o_sc);

  auto cdiv = [](int a, int b) { return (a + b - 1) / b; };
  pack_w<<<cdiv(96 * 192, 256), 256, 0, stream>>>(proj_w, projw_p, 96, 192);
  pack_w<<<cdiv(96 * 576, 256), 256, 0, stream>>>(qkv_w, qkvw_p, 96, 576);
  pack_w<<<cdiv(192 * 192, 256), 256, 0, stream>>>(aproj_w, attnw_p, 192, 192);
  pack_w<<<cdiv(192 * 768, 256), 256, 0, stream>>>(mlp_w1, w1p, 192, 768);
  pack_w<<<cdiv(768 * 192, 256), 256, 0, stream>>>(mlp_w2, w2p, 768, 192);

  k_ln_qkv<<<NWINDOWS, 256, 0, stream>>>(x, n1g, n1b, qkvw_p, qkv_b, projw_p,
                                         proj_b, qp_ws, k_ws, v_ws, sc_ws);
  k_attn<<<NWINDOWS, 256, 0, stream>>>(qp_ws, k_ws, v_ws, sc_ws, attnw_p,
                                       aproj_b, y);
  k_mlp<<<(HP * WP * BATCH) / 64, 256, 0, stream>>>(n2g, n2b, w1p, mlp_b1, w2p,
                                                    mlp_b2, y);
}
